// GraphReasonedLVM_71794673320519
// MI455X (gfx1250) — matrix operand order repaired and checked by reading, not compiled, not run on hardware
//
#include <hip/hip_runtime.h>
#include <hip/hip_bf16.h>

// ---------------------------------------------------------------------------
// GraphReasonedLVM for gfx1250 (MI455X): bf16 WMMA GEMMs + tridiagonal GAT.
//   - All weights pre-converted to bf16 and pre-transposed to [N][K]
//   - All GEMM activations kept in bf16 (pooled, h, cat)
//   - GEMM: 2 waves/block, 16x64 per wave, LDS-staged B^T tile (async-to-LDS)
// ---------------------------------------------------------------------------

#define HID   512
#define HEADS 8
#define N_LM  543
#define T_FR  32
#define ROWS  (T_FR * N_LM)        // 17376 == 1086*16
#define FCK   200704               // 64*16*14*14
#define CATW  (HEADS * HID)        // 4096

typedef __attribute__((ext_vector_type(16))) __bf16          v16bf;
typedef __attribute__((ext_vector_type(16))) unsigned short  v16u16;
typedef __attribute__((ext_vector_type(8)))  float           v8f;
typedef __attribute__((ext_vector_type(8)))  unsigned int    v8u;
typedef __attribute__((ext_vector_type(4)))  int             v4i;
// AS-qualified vector types for the async-to-LDS builtin
// (probe: param0 is 'int4 __device__*' == AS1; LDS side is AS3)
typedef __attribute__((address_space(1))) v4i as1_v4i;
typedef __attribute__((address_space(3))) v4i as3_v4i;

#if defined(__HIP_DEVICE_COMPILE__) && \
    __has_builtin(__builtin_amdgcn_global_load_async_to_lds_b128) && \
    __has_builtin(__builtin_amdgcn_s_wait_asynccnt)
  #define GEMM_ASYNC 1
#else
  #define GEMM_ASYNC 0
  #if defined(__HIP_DEVICE_COMPILE__)
    #warning "gfx1250 async-to-LDS builtins unavailable; using sync LDS staging"
  #endif
#endif

__device__ __forceinline__ unsigned short f32_to_bf16(float f) {
    unsigned u = __builtin_bit_cast(unsigned, f);
    u += 0x7FFFu + ((u >> 16) & 1u);           // round-to-nearest-even
    return (unsigned short)(u >> 16);
}
__device__ __forceinline__ float bf16_to_f32(unsigned short u) {
    return __builtin_bit_cast(float, ((unsigned)u) << 16);
}

// ---------------------------------------------------------------------------
// Kernel 0: convert + transpose weights  Bt[n*K + k] = bf16(B[k*N + n])
// ---------------------------------------------------------------------------
__global__ void cvt_transpose_bf16_kernel(const float* __restrict__ B,
                                          unsigned short* __restrict__ Bt,
                                          int Kd, int Nd) {
    const long long idx = (long long)blockIdx.x * blockDim.x + threadIdx.x;
    if (idx >= (long long)Kd * Nd) return;
    const int k = (int)(idx / Nd);
    const int n = (int)(idx - (long long)k * Nd);
    Bt[(size_t)n * Kd + k] = f32_to_bf16(B[idx]);
}

// ---------------------------------------------------------------------------
// Kernel 1: direct Conv3d (stride 1,2,2; pad 1,3,3) + ReLU + 4x4 avg-pool
// out: bf16 [32, 64*16*14*14] in fc order
// ---------------------------------------------------------------------------
__global__ __launch_bounds__(256)
void conv_pool_kernel(const float* __restrict__ in, const float* __restrict__ w,
                      const float* __restrict__ bias, unsigned short* __restrict__ out) {
    __shared__ float wsm[441];
    const int od = blockIdx.x, c = blockIdx.y, b = blockIdx.z;
    for (int i = threadIdx.x; i < 441; i += blockDim.x) wsm[i] = w[c * 441 + i];
    __syncthreads();
    const int t = threadIdx.x;
    if (t >= 196) return;
    const int ph = t / 14, pw = t % 14;
    const float bc = bias[c];
    float s = 0.f;
    for (int oh = ph * 4; oh < ph * 4 + 4; ++oh) {
        for (int ow = pw * 4; ow < pw * 4 + 4; ++ow) {
            float acc = bc;
            for (int ic = 0; ic < 3; ++ic) {
                for (int kd = 0; kd < 3; ++kd) {
                    const int id = od - 1 + kd;
                    if (id < 0 || id >= 16) continue;
                    const float* iplane = in + (((size_t)b * 3 + ic) * 16 + id) * 112 * 112;
                    for (int kh = 0; kh < 7; ++kh) {
                        const int ih = oh * 2 - 3 + kh;
                        if (ih < 0 || ih >= 112) continue;
                        const float* irow = iplane + (size_t)ih * 112;
                        const float* wrow = wsm + ((ic * 3 + kd) * 7 + kh) * 7;
                        #pragma unroll
                        for (int kw = 0; kw < 7; ++kw) {
                            const int iw = ow * 2 - 3 + kw;
                            if (iw < 0 || iw >= 112) continue;
                            acc = fmaf(irow[iw], wrow[kw], acc);
                        }
                    }
                }
            }
            s += fmaxf(acc, 0.f);
        }
    }
    out[(size_t)b * FCK + ((c * 16 + od) * 196 + ph * 14 + pw)] = f32_to_bf16(s * (1.f / 16.f));
}

// ---------------------------------------------------------------------------
// Kernel 2: WMMA GEMM  C[M,N] = act(A[M,K] @ B[K,N] + bias)
//   A : bf16 row-major [M,K], lda
//   Bt: bf16 col-major [N][K] (pre-transposed weights)
//   C : f32 row-major, ldc = N
// block = 64 threads (2 waves); wave w computes rows (blockIdx.y*2+w)*16..+15,
// cols blockIdx.x*64..+63 (4 WMMA accumulators). B^T tile (64 cols x 32 K,
// LDS stride 40 ushorts vs 32 to break bank conflicts) staged per K-step.
// M % 32 == 0, N % 64 == 0, K % 32 == 0.
// ---------------------------------------------------------------------------
__global__ __launch_bounds__(64)
void wmma_gemm_kernel(const unsigned short* __restrict__ A, int lda,
                      const unsigned short* __restrict__ Bt,
                      float* __restrict__ C, int ldc, int K,
                      const float* __restrict__ bias, int act) {
    __shared__ unsigned short ldsBt[64 * 40];    // [col][40] (32 used + 8 pad)
    const int tid  = threadIdx.x;
    const int lane = tid & 31;
    const int wid  = tid >> 5;
    const int hf   = lane >> 4;      // 16-lane half
    const int lidx = lane & 15;      // m for A / n for B,C
    const int n0 = blockIdx.x * 64;
    const int m0 = (blockIdx.y * 2 + wid) * 16;

    // staging: 256 chunks of 16B (8 bf16 along K); 4 chunks per thread
    // chunk idx = i*64 + tid -> col = idx>>2, part = idx&3
    const int part = tid & 3;
    const unsigned short* gB[4];
    unsigned short*       lB[4];
    #pragma unroll
    for (int i = 0; i < 4; ++i) {
        const int col = i * 16 + (tid >> 2);
        gB[i] = Bt + (size_t)(n0 + col) * K + part * 8;
        lB[i] = ldsBt + col * 40 + part * 8;
    }

    const unsigned short* aRow = A + (size_t)(m0 + lidx) * lda + hf * 8;

    v8f acc[4];
    #pragma unroll
    for (int t = 0; t < 4; ++t) acc[t] = (v8f){0.f,0.f,0.f,0.f,0.f,0.f,0.f,0.f};

    for (int k0 = 0; k0 < K; k0 += 32) {
#if GEMM_ASYNC
        #pragma unroll
        for (int i = 0; i < 4; ++i) {
            __builtin_amdgcn_global_load_async_to_lds_b128(
                (as1_v4i*)gB[i], (as3_v4i*)lB[i], 0, 0);
            gB[i] += 32;
        }
        __builtin_amdgcn_s_wait_asynccnt(0);
#else
        #pragma unroll
        for (int i = 0; i < 4; ++i) {
            const uint4 v = *(const uint4*)gB[i];
            *(uint4*)lB[i] = v;
            gB[i] += 32;
        }
#endif
        __syncthreads();

        if (k0 + 32 < K) {  // global_prefetch_b8 hints for next chunks
            __builtin_prefetch(aRow + 32, 0, 3);
            __builtin_prefetch(gB[0], 0, 3);
        }

        // ---- A fragment (16-bit A 16x32 layout: j<8 -> K=hf*8+j ; j>=8 -> 16+hf*8+j-8)
        const uint4 alo = *(const uint4*)(aRow);
        const uint4 ahi = *(const uint4*)(aRow + 16);
        aRow += 32;
        v8u aw;
        aw[0] = alo.x; aw[1] = alo.y; aw[2] = alo.z; aw[3] = alo.w;
        aw[4] = ahi.x; aw[5] = ahi.y; aw[6] = ahi.z; aw[7] = ahi.w;
        const v16bf afrag = __builtin_bit_cast(v16bf, aw);

        // ---- 4 B fragments + WMMAs (B layout: e<8 -> K = hf*8 + e ; e>=8 -> 16+hf*8+e-8, col = n)
        #pragma unroll
        for (int t = 0; t < 4; ++t) {
            const unsigned short* lptr = ldsBt + (t * 16 + lidx) * 40 + hf * 8;
            const uint4 q0 = *(const uint4*)(lptr);
            const uint4 q1 = *(const uint4*)(lptr + 16);
            v8u bw;
            bw[0] = q0.x; bw[1] = q0.y; bw[2] = q0.z; bw[3] = q0.w;
            bw[4] = q1.x; bw[5] = q1.y; bw[6] = q1.z; bw[7] = q1.w;
            acc[t] = __builtin_amdgcn_wmma_f32_16x16x32_bf16(
                         false, afrag, false, __builtin_bit_cast(v16bf, bw),
                         (short)0, acc[t], false, false);
        }
        __syncthreads();
    }

    // ---- epilogue: C/D 16x16 f32: VGPR r -> M = 8*hf + r, N = lidx ----
    #pragma unroll
    for (int t = 0; t < 4; ++t) {
        const int n = n0 + t * 16 + lidx;
        float bv = bias ? bias[n] : 0.f;
        #pragma unroll
        for (int r = 0; r < 8; ++r) {
            const int m = m0 + hf * 8 + r;
            float v = acc[t][r] + bv;
            if (act == 1) v = fmaxf(v, 0.f);
            C[(size_t)m * ldc + n] = v;
        }
    }
}

// ---------------------------------------------------------------------------
// Kernel 3: pose linear  h[r,c] = bf16(pose[r,:3] @ lin_w[:, c] + lin_b[c])
// ---------------------------------------------------------------------------
__global__ void pose_lin_kernel(const float* __restrict__ pose,
                                const float* __restrict__ lw,
                                const float* __restrict__ lb,
                                unsigned short* __restrict__ h) {
    const int idx = blockIdx.x * blockDim.x + threadIdx.x;
    if (idx >= ROWS * HID) return;
    const int r = idx >> 9, c = idx & 511;
    const float* p = pose + (size_t)r * 3;
    h[idx] = f32_to_bf16(lb[c] + p[0] * lw[c] + p[1] * lw[HID + c] + p[2] * lw[2 * HID + c]);
}

// ---------------------------------------------------------------------------
// Kernel 4: dual row-dot  src[r] = Wh[r,:]·a_src,  dst[r] = Wh[r,:]·a_dst
// ---------------------------------------------------------------------------
__global__ __launch_bounds__(32)
void rowdot2_kernel(const float* __restrict__ Wh, const float* __restrict__ a_src,
                    const float* __restrict__ a_dst, float* __restrict__ src,
                    float* __restrict__ dst) {
    const int r = blockIdx.x;
    const int lane = threadIdx.x;
    const float* row = Wh + (size_t)r * HID;
    float s = 0.f, d = 0.f;
    #pragma unroll
    for (int k = lane; k < HID; k += 32) {
        const float v = row[k];
        s = fmaf(v, a_src[k], s);
        d = fmaf(v, a_dst[k], d);
    }
    #pragma unroll
    for (int off = 16; off; off >>= 1) {
        s += __shfl_down(s, off, 32);
        d += __shfl_down(d, off, 32);
    }
    if (lane == 0) { src[r] = s; dst[r] = d; }
}

// ---------------------------------------------------------------------------
// Kernel 5: tridiagonal GAT attention + ELU -> bf16 output
//   double_elu=0: single ELU (head outputs into cat columns)
//   double_elu=1: elu(elu(x)) (reference applies elu twice on the out-GAT)
// ---------------------------------------------------------------------------
__global__ __launch_bounds__(256)
void gat_attn_kernel(const float* __restrict__ Wh, const float* __restrict__ src,
                     const float* __restrict__ dst, unsigned short* __restrict__ out,
                     int ldd, int col_off, int double_elu) {
    const int r = blockIdx.x;                 // global row 0..ROWS-1
    const int f = r / N_LM;
    const int i = r - f * N_LM;
    const int base = f * N_LM;

    const float sr = src[r];
    float e[3];
    float m = -1e30f;
    #pragma unroll
    for (int jj = 0; jj < 3; ++jj) {
        const int j = i - 1 + jj;
        if (j < 0 || j >= N_LM) { e[jj] = -1e30f; continue; }
        float x = sr + dst[base + j];
        x = (x >= 0.f) ? x : 0.2f * x;        // leaky_relu(0.2)
        e[jj] = x;
        m = fmaxf(m, x);
    }
    float wgt[3], ssum = 0.f;
    #pragma unroll
    for (int jj = 0; jj < 3; ++jj) {
        const float w = (e[jj] > -1e29f) ? __expf(e[jj] - m) : 0.f;
        wgt[jj] = w; ssum += w;
    }
    const float inv = 1.f / ssum;
    const float w0 = wgt[0] * inv, w1 = wgt[1] * inv, w2 = wgt[2] * inv;

    for (int c = threadIdx.x; c < HID; c += blockDim.x) {
        float acc = w1 * Wh[(size_t)r * HID + c];
        if (i > 0)         acc = fmaf(w0, Wh[(size_t)(r - 1) * HID + c], acc);
        if (i < N_LM - 1)  acc = fmaf(w2, Wh[(size_t)(r + 1) * HID + c], acc);
        float o = (acc > 0.f) ? acc : (__expf(acc) - 1.f);          // elu
        if (double_elu) o = (o > 0.f) ? o : (__expf(o) - 1.f);      // elu(elu)
        out[(size_t)r * ldd + col_off + c] = f32_to_bf16(o);
    }
}

// ---------------------------------------------------------------------------
// Kernel 6: per-frame mean over landmarks (bf16 in, f32 out)
// ---------------------------------------------------------------------------
__global__ void pose_mean_kernel(const unsigned short* __restrict__ h, float* __restrict__ pf) {
    const int idx = blockIdx.x * blockDim.x + threadIdx.x;
    if (idx >= T_FR * HID) return;
    const int b = idx >> 9, c = idx & 511;
    const unsigned short* p = h + (size_t)b * N_LM * HID + c;
    float s = 0.f;
    for (int i = 0; i < N_LM; ++i) s += bf16_to_f32(p[(size_t)i * HID]);
    pf[idx] = s * (1.f / (float)N_LM);
}

// ---------------------------------------------------------------------------
// Kernel 7: small dense matmul  C[r,c] = A[r,:K] @ B (+bias)
//   transB=0: B[k*ldb+c]   transB=1: B[c*ldb+k]
// ---------------------------------------------------------------------------
__global__ void small_mm_kernel(const float* __restrict__ A, int lda,
                                const float* __restrict__ B, int ldb, int transB,
                                const float* __restrict__ bias,
                                float* __restrict__ C, int rows, int cols, int K) {
    const int idx = blockIdx.x * blockDim.x + threadIdx.x;
    if (idx >= rows * cols) return;
    const int r = idx / cols, c = idx - r * cols;
    float s = bias ? bias[c] : 0.f;
    const float* ar = A + (size_t)r * lda;
    if (transB) {
        const float* br = B + (size_t)c * ldb;
        for (int k = 0; k < K; ++k) s = fmaf(ar[k], br[k], s);
    } else {
        for (int k = 0; k < K; ++k) s = fmaf(ar[k], B[(size_t)k * ldb + c], s);
    }
    C[(size_t)r * cols + c] = s;
}

// ---------------------------------------------------------------------------
// Kernel 8: in-place row softmax over 512 cols, one wave per row
// ---------------------------------------------------------------------------
__global__ __launch_bounds__(32)
void row_softmax_kernel(float* __restrict__ t) {
    const int r = blockIdx.x;
    const int lane = threadIdx.x;
    float* row = t + (size_t)r * HID;
    float vals[16];
    float m = -1e30f;
    #pragma unroll
    for (int j = 0; j < 16; ++j) { vals[j] = row[j * 32 + lane]; m = fmaxf(m, vals[j]); }
    #pragma unroll
    for (int off = 16; off; off >>= 1) m = fmaxf(m, __shfl_xor(m, off, 32));
    float s = 0.f;
    #pragma unroll
    for (int j = 0; j < 16; ++j) { vals[j] = __expf(vals[j] - m); s += vals[j]; }
    #pragma unroll
    for (int off = 16; off; off >>= 1) s += __shfl_xor(s, off, 32);
    const float inv = 1.f / s;
    #pragma unroll
    for (int j = 0; j < 16; ++j) row[j * 32 + lane] = vals[j] * inv;
}

// ---------------------------------------------------------------------------
// Kernel 9: LoRA  adapted = constrained + 0.1 * (u @ lora_B),  u: [32,16]
// ---------------------------------------------------------------------------
__global__ void lora_kernel(const float* __restrict__ cons, const float* __restrict__ u,
                            const float* __restrict__ loraB, float* __restrict__ adapted) {
    const int idx = blockIdx.x * blockDim.x + threadIdx.x;
    if (idx >= T_FR * HID) return;
    const int b = idx >> 9, n = idx & 511;
    float s = 0.f;
    #pragma unroll
    for (int r = 0; r < 16; ++r) s = fmaf(u[b * 16 + r], loraB[r * HID + n], s);
    adapted[idx] = cons[idx] + 0.1f * s;
}

// ---------------------------------------------------------------------------
// Kernel 10: output heads -> d_out (ctc 448 | slot 320 | mean 32 | var 32 | abstain 32)
// ---------------------------------------------------------------------------
__global__ void heads_kernel(const float* __restrict__ adapted,
                             const float* __restrict__ ctc_w, const float* __restrict__ ctc_b,
                             const float* __restrict__ slot_w, const float* __restrict__ slot_b,
                             const float* __restrict__ mean_w, const float* __restrict__ mean_b,
                             const float* __restrict__ var_w, const float* __restrict__ var_b,
                             float* __restrict__ out) {
    const int idx = blockIdx.x * blockDim.x + threadIdx.x;
    if (idx >= T_FR * 26) return;
    const int b = idx / 26, c = idx - b * 26;
    const float* a = adapted + (size_t)b * HID;
    if (c < 14) {
        float s = ctc_b[c];
        for (int k = 0; k < HID; ++k) s = fmaf(a[k], ctc_w[k * 14 + c], s);
        out[b * 14 + c] = s;
    } else if (c < 24) {
        const int g = c - 14;
        float s = slot_b[g];
        for (int k = 0; k < HID; ++k) s = fmaf(a[k], slot_w[k * 10 + g], s);
        out[448 + b * 10 + g] = s;
    } else if (c == 24) {
        float s = mean_b[0];
        for (int k = 0; k < HID; ++k) s = fmaf(a[k], mean_w[k], s);
        const float mv = 1.f / (1.f + __expf(-s));
        out[768 + b] = mv;
        out[832 + b] = (mv < 0.7f) ? 1.f : 0.f;    // abstain
    } else {
        float s = var_b[0];
        for (int k = 0; k < HID; ++k) s = fmaf(a[k], var_w[k], s);
        out[800 + b] = fmaxf(s, 0.f) + __logf(1.f + __expf(-fabsf(s)));  // softplus
    }
}

// ---------------------------------------------------------------------------
// Host launcher
// ---------------------------------------------------------------------------
extern "C" void kernel_launch(void* const* d_in, const int* in_sizes, int n_in,
                              void* d_out, int out_size, void* d_ws, size_t ws_size,
                              hipStream_t stream) {
    (void)in_sizes; (void)n_in; (void)out_size; (void)ws_size;
    const float* rgb       = (const float*)d_in[0];
    const float* pose      = (const float*)d_in[1];
    const float* conv_w    = (const float*)d_in[2];
    const float* conv_b    = (const float*)d_in[3];
    const float* fc_w      = (const float*)d_in[4];
    const float* fc_b      = (const float*)d_in[5];
    const float* lin_w     = (const float*)d_in[6];
    const float* lin_b     = (const float*)d_in[7];
    const float* W1        = (const float*)d_in[8];
    const float* a1        = (const float*)d_in[9];
    const float* W1o       = (const float*)d_in[10];
    const float* a1o       = (const float*)d_in[11];
    const float* W2        = (const float*)d_in[12];
    const float* a2        = (const float*)d_in[13];
    const float* W2o       = (const float*)d_in[14];
    const float* a2o       = (const float*)d_in[15];
    const float* in_proj_w = (const float*)d_in[16];
    const float* in_proj_b = (const float*)d_in[17];
    const float* out_w     = (const float*)d_in[18];
    const float* out_b     = (const float*)d_in[19];
    const float* factor_w  = (const float*)d_in[20];
    const float* lora_A    = (const float*)d_in[21];
    const float* lora_B    = (const float*)d_in[22];
    const float* ctc_w     = (const float*)d_in[23];
    const float* ctc_b     = (const float*)d_in[24];
    const float* slot_w    = (const float*)d_in[25];
    const float* slot_b    = (const float*)d_in[26];
    const float* mean_w    = (const float*)d_in[27];
    const float* mean_b    = (const float*)d_in[28];
    const float* var_w     = (const float*)d_in[29];
    const float* var_b     = (const float*)d_in[30];

    // ---- bump allocator over d_ws ----
    char* wsp = (char*)d_ws;
    auto alloc = [&](size_t bytes) -> void* {
        void* p = (void*)wsp;
        wsp += (bytes + 255) & ~(size_t)255;
        return p;
    };
    unsigned short* fcwt     = (unsigned short*)alloc((size_t)FCK * HID * 2);   // 205 MB
    unsigned short* W1t      = (unsigned short*)alloc((size_t)HEADS * HID * HID * 2);
    unsigned short* W2t      = (unsigned short*)alloc((size_t)HEADS * HID * HID * 2);
    unsigned short* W1ot     = (unsigned short*)alloc((size_t)CATW * HID * 2);
    unsigned short* W2ot     = (unsigned short*)alloc((size_t)CATW * HID * 2);
    unsigned short* pooledb  = (unsigned short*)alloc((size_t)T_FR * FCK * 2);  // 12.8 MB
    float*          rgb_feat = (float*)alloc((size_t)T_FR * HID * 4);
    unsigned short* h_a      = (unsigned short*)alloc((size_t)ROWS * HID * 2);  // 17.8 MB
    unsigned short* h_b      = (unsigned short*)alloc((size_t)ROWS * HID * 2);  // 17.8 MB
    float*          Wh       = (float*)alloc((size_t)ROWS * HID * 4);           // 35.6 MB
    unsigned short* catb     = (unsigned short*)alloc((size_t)ROWS * CATW * 2); // 142 MB
    float*          srcb     = (float*)alloc((size_t)ROWS * 4);
    float*          dstb     = (float*)alloc((size_t)ROWS * 4);
    float*          posefeat = (float*)alloc((size_t)T_FR * HID * 4);
    float*          vbuf     = (float*)alloc((size_t)T_FR * HID * 4);
    float*          fusedb   = (float*)alloc((size_t)T_FR * HID * 4);
    float*          tbuf     = (float*)alloc((size_t)T_FR * HID * 4);
    float*          ubuf     = (float*)alloc((size_t)T_FR * 16 * 4);
    float*          adapted  = (float*)alloc((size_t)T_FR * HID * 4);

    // ---- weight precondition: f32 row-major -> bf16 col-major ----
    {
        const long long nfc = (long long)FCK * HID;
        cvt_transpose_bf16_kernel<<<(unsigned)((nfc + 255) / 256), 256, 0, stream>>>(
            fc_w, fcwt, FCK, HID);
        const int nh = HID * HID;
        for (int hd = 0; hd < HEADS; ++hd) {
            cvt_transpose_bf16_kernel<<<(nh + 255) / 256, 256, 0, stream>>>(
                W1 + (size_t)hd * nh, W1t + (size_t)hd * nh, HID, HID);
            cvt_transpose_bf16_kernel<<<(nh + 255) / 256, 256, 0, stream>>>(
                W2 + (size_t)hd * nh, W2t + (size_t)hd * nh, HID, HID);
        }
        const int no = CATW * HID;
        cvt_transpose_bf16_kernel<<<(no + 255) / 256, 256, 0, stream>>>(W1o, W1ot, CATW, HID);
        cvt_transpose_bf16_kernel<<<(no + 255) / 256, 256, 0, stream>>>(W2o, W2ot, CATW, HID);
    }

    // ---- RGB stream: conv+pool (bf16 out), then WMMA FC ----
    conv_pool_kernel<<<dim3(16, 64, 32), 256, 0, stream>>>(rgb, conv_w, conv_b, pooledb);
    wmma_gemm_kernel<<<dim3(HID / 64, T_FR / 32), 64, 0, stream>>>(
        pooledb, FCK, fcwt, rgb_feat, HID, FCK, fc_b, 1);

    // ---- Pose stream: linear embed, then 2 multi-head GAT layers ----
    pose_lin_kernel<<<(ROWS * HID + 255) / 256, 256, 0, stream>>>(pose, lin_w, lin_b, h_a);

    unsigned short* h_cur = h_a;
    unsigned short* h_nxt = h_b;
    for (int layer = 0; layer < 2; ++layer) {
        const unsigned short* Wt  = (layer == 0) ? W1t  : W2t;
        const float*          aa  = (layer == 0) ? a1   : a2;
        const unsigned short* Wot = (layer == 0) ? W1ot : W2ot;
        const float*          ao  = (layer == 0) ? a1o  : a2o;

        for (int head = 0; head < HEADS; ++head) {
            // Wh = h @ W[head]   (WMMA, M=17376, N=512, K=512)
            wmma_gemm_kernel<<<dim3(HID / 64, ROWS / 32), 64, 0, stream>>>(
                h_cur, HID, Wt + (size_t)head * HID * HID, Wh, HID, HID, nullptr, 0);
            rowdot2_kernel<<<ROWS, 32, 0, stream>>>(
                Wh, aa + (size_t)head * 2 * HID, aa + (size_t)head * 2 * HID + HID,
                srcb, dstb);
            // tridiagonal softmax-attn + elu -> bf16 column block of cat
            gat_attn_kernel<<<ROWS, 256, 0, stream>>>(
                Wh, srcb, dstb, catb, CATW, head * HID, 0);
        }
        // output GAT: Wh_o = cat @ Wo  (WMMA, K=4096)
        wmma_gemm_kernel<<<dim3(HID / 64, ROWS / 32), 64, 0, stream>>>(
            catb, CATW, Wot, Wh, HID, CATW, nullptr, 0);
        rowdot2_kernel<<<ROWS, 32, 0, stream>>>(Wh, ao, ao + HID, srcb, dstb);
        gat_attn_kernel<<<ROWS, 256, 0, stream>>>(Wh, srcb, dstb, h_nxt, HID, 0, 1);

        unsigned short* tmp = h_cur; h_cur = h_nxt; h_nxt = tmp;
    }
    pose_mean_kernel<<<(T_FR * HID + 255) / 256, 256, 0, stream>>>(h_cur, posefeat);

    // ---- Fusion: softmax over S=1 is exactly 1 => fused = v @ out_w.T + out_b ----
    small_mm_kernel<<<(T_FR * HID + 255) / 256, 256, 0, stream>>>(
        posefeat, HID, in_proj_w + (size_t)2 * HID * HID, HID, 1,
        in_proj_b + 2 * HID, vbuf, T_FR, HID, HID);
    small_mm_kernel<<<(T_FR * HID + 255) / 256, 256, 0, stream>>>(
        vbuf, HID, out_w, HID, 1, out_b, fusedb, T_FR, HID, HID);

    // ---- FactorGraph softmax + LoRA ----
    small_mm_kernel<<<(T_FR * HID + 255) / 256, 256, 0, stream>>>(
        fusedb, HID, factor_w, HID, 0, nullptr, tbuf, T_FR, HID, HID);
    row_softmax_kernel<<<T_FR, 32, 0, stream>>>(tbuf);                     // constrained
    small_mm_kernel<<<(T_FR * 16 + 255) / 256, 256, 0, stream>>>(
        tbuf, HID, lora_A, 16, 0, nullptr, ubuf, T_FR, 16, HID);
    lora_kernel<<<(T_FR * HID + 255) / 256, 256, 0, stream>>>(tbuf, ubuf, lora_B, adapted);

    // ---- Heads ----
    heads_kernel<<<(T_FR * 26 + 255) / 256, 256, 0, stream>>>(
        adapted, ctc_w, ctc_b, slot_w, slot_b, mean_w, mean_b, var_w, var_b,
        (float*)d_out);
}
